// GATEncoder_18803366822141
// MI455X (gfx1250) — matrix operand order repaired and checked by reading, not compiled, not run on hardware
//
#include <hip/hip_runtime.h>
#include <hip/hip_bf16.h>

typedef __attribute__((ext_vector_type(16))) _Float16 v16h;
typedef __attribute__((ext_vector_type(8)))  float    v8f;

#define IN_CH   128
#define HID     32
#define HEADS   2
#define C1      (HEADS * HID)   /* 64 */
#define OUT_CH  128
#define NEG_SLOPE 0.2f

// ---------------------------------------------------------------------------
// ordered-uint encoding for float atomicMax (identity element == 0u)
// ---------------------------------------------------------------------------
__device__ inline unsigned fenc(float f) {
    unsigned u = __float_as_uint(f);
    return (u & 0x80000000u) ? ~u : (u | 0x80000000u);
}
__device__ inline float fdec(unsigned u) {
    return __uint_as_float((u & 0x80000000u) ? (u ^ 0x80000000u) : ~u);
}
__device__ inline float leaky(float v) { return v > 0.0f ? v : NEG_SLOPE * v; }
__device__ inline float elu(float v)   { return v > 0.0f ? v : (expf(v) - 1.0f); }

__device__ inline void edge_sd(const int* __restrict__ ei, int e, int nE,
                               int& s, int& d) {
    if (e < nE) { s = ei[e]; d = ei[nE + e]; }
    else        { s = d = e - nE; }          // self loop
}

// ---------------------------------------------------------------------------
// Weight convert + transpose:  W[K][N] f32  ->  Wt[N][K] f16
// ---------------------------------------------------------------------------
__global__ void conv_w_kernel(const float* __restrict__ W1, const float* __restrict__ W2,
                              _Float16* __restrict__ wt1, _Float16* __restrict__ wt2) {
    int i = blockIdx.x * blockDim.x + threadIdx.x;
    if (i < C1 * IN_CH) {            // wt1: [64][128]
        int n = i / IN_CH, k = i % IN_CH;
        wt1[i] = (_Float16)W1[k * C1 + n];
    }
    if (i < OUT_CH * C1) {           // wt2: [128][64]
        int n = i / C1, k = i % C1;
        wt2[i] = (_Float16)W2[k * OUT_CH + n];
    }
}

// ---------------------------------------------------------------------------
// WMMA GEMM:  C[M][N] = A[M][K](f32) * Wt[N][K](f16, pre-transposed)
// block = 128 threads (4 waves); each block does a 64-row M tile, full N.
// ---------------------------------------------------------------------------
template<int K, int N>
__global__ __launch_bounds__(128)
void gemm_wmma(const float* __restrict__ A, const _Float16* __restrict__ Bt,
               float* __restrict__ C, int M) {
    __shared__ __align__(16) _Float16 sA[64 * K];   // row-major [64][K]
    __shared__ __align__(16) _Float16 sB[N * K];    // N-major  [N][K]

    const int tid  = threadIdx.x;
    const int lane = tid & 31;
    const int wave = tid >> 5;
    const int row0 = blockIdx.x * 64;

    __builtin_prefetch(Bt, 0, 0);

    // stage Bt (whole weight matrix), 16B vector copies
    for (int i = tid; i < (N * K) / 8; i += 128)
        ((float4*)sB)[i] = ((const float4*)Bt)[i];

    // stage A tile: f32 global -> f16 LDS (zero-pad rows >= M)
    for (int i = tid; i < (64 * K) / 4; i += 128) {
        int r  = i / (K / 4);
        int c4 = i % (K / 4);
        int gr = row0 + r;
        float4 v = (gr < M) ? ((const float4*)(A + (size_t)gr * K))[c4]
                            : make_float4(0.f, 0.f, 0.f, 0.f);
        _Float16 h4[4] = { (_Float16)v.x, (_Float16)v.y, (_Float16)v.z, (_Float16)v.w };
        *(float2*)(&sA[r * K + 4 * c4]) = *(float2*)h4;
    }
    __syncthreads();

    const int m_lane = lane & 15;            // N column within tile / A row low bits
    const int hi     = lane >> 4;            // 0 or 1
    const int a_row  = wave * 16 + m_lane;   // A row within 64-row tile
    const int kb     = hi * 8;               // A K sub-offset
    const int koffB  = hi * 8;               // B K sub-offset

    v8f acc[N / 16];
    #pragma unroll
    for (int nt = 0; nt < N / 16; ++nt) {
        v8f z = {0.f, 0.f, 0.f, 0.f, 0.f, 0.f, 0.f, 0.f};
        acc[nt] = z;
    }

    #pragma unroll
    for (int k0 = 0; k0 < K; k0 += 32) {
        v16h a;
        *((float4*)&a)     = *(const float4*)&sA[a_row * K + k0 + kb];
        *((float4*)&a + 1) = *(const float4*)&sA[a_row * K + k0 + 16 + kb];
        #pragma unroll
        for (int nt = 0; nt < N / 16; ++nt) {
            const _Float16* bp = &sB[(nt * 16 + m_lane) * K + k0 + koffB];
            v16h b;
            *((float4*)&b)     = *(const float4*)bp;
            *((float4*)&b + 1) = *(const float4*)(bp + 16);
            acc[nt] = __builtin_amdgcn_wmma_f32_16x16x32_f16(
                          false, a, false, b, (short)0, acc[nt], false, false);
        }
    }

    // store D: VGPR r <-> M = r + 8*hi, N = lane&15
    #pragma unroll
    for (int nt = 0; nt < N / 16; ++nt) {
        int n_g = nt * 16 + m_lane;
        #pragma unroll
        for (int r = 0; r < 8; ++r) {
            int gr = row0 + wave * 16 + hi * 8 + r;
            if (gr < M) C[(size_t)gr * N + n_g] = acc[nt][r];
        }
    }
}

// ---------------------------------------------------------------------------
// Per-node attention coefficients: a_s[n][h] = sum_d xl[n][h*D+d]*att_s[h*D+d]
// one wave per node
// ---------------------------------------------------------------------------
__global__ void attn_coef(const float* __restrict__ xl,
                          const float* __restrict__ atts, const float* __restrict__ attd,
                          float* __restrict__ as_, float* __restrict__ ad_,
                          int nN, int H, int D) {
    int node = blockIdx.x * (blockDim.x >> 5) + (threadIdx.x >> 5);
    int lane = threadIdx.x & 31;
    if (node >= nN) return;
    int HD = H * D;
    float aS[2] = {0.f, 0.f}, aD[2] = {0.f, 0.f};
    for (int j = lane; j < HD; j += 32) {
        int h = j / D;
        float v = xl[(size_t)node * HD + j];
        aS[h] += v * atts[j];
        aD[h] += v * attd[j];
    }
    for (int off = 16; off; off >>= 1) {
        aS[0] += __shfl_xor(aS[0], off);
        aS[1] += __shfl_xor(aS[1], off);
        aD[0] += __shfl_xor(aD[0], off);
        aD[1] += __shfl_xor(aD[1], off);
    }
    if (lane == 0)
        for (int h = 0; h < H; ++h) {
            as_[(size_t)node * H + h] = aS[h];
            ad_[(size_t)node * H + h] = aD[h];
        }
}

// ---------------------------------------------------------------------------
// Edge pass 1: segment max over dst (ordered-uint atomicMax)
// ---------------------------------------------------------------------------
__global__ void edge_max_kernel(const int* __restrict__ ei,
                                const float* __restrict__ as_, const float* __restrict__ ad_,
                                unsigned* __restrict__ mx, int nE, int nN, int H) {
    int e = blockIdx.x * blockDim.x + threadIdx.x;
    if (e >= nE + nN) return;
    int s, d; edge_sd(ei, e, nE, s, d);
    for (int h = 0; h < H; ++h) {
        float v = leaky(as_[(size_t)s * H + h] + ad_[(size_t)d * H + h]);
        atomicMax(&mx[(size_t)d * H + h], fenc(v));
    }
}

// ---------------------------------------------------------------------------
// Edge pass 2: denom = segment_sum(exp(e - max))
// ---------------------------------------------------------------------------
__global__ void edge_den_kernel(const int* __restrict__ ei,
                                const float* __restrict__ as_, const float* __restrict__ ad_,
                                const unsigned* __restrict__ mx, float* __restrict__ den,
                                int nE, int nN, int H) {
    int e = blockIdx.x * blockDim.x + threadIdx.x;
    if (e >= nE + nN) return;
    int s, d; edge_sd(ei, e, nE, s, d);
    for (int h = 0; h < H; ++h) {
        float v = leaky(as_[(size_t)s * H + h] + ad_[(size_t)d * H + h]);
        float m = fdec(mx[(size_t)d * H + h]);
        atomicAdd(&den[(size_t)d * H + h], expf(v - m));
    }
}

// ---------------------------------------------------------------------------
// Edge pass 3: out[dst] += alpha * xl[src]   (wave per edge)
// ---------------------------------------------------------------------------
__global__ void edge_scatter_kernel(const int* __restrict__ ei,
                                    const float* __restrict__ as_, const float* __restrict__ ad_,
                                    const unsigned* __restrict__ mx, const float* __restrict__ den,
                                    const float* __restrict__ xl, float* __restrict__ out,
                                    int nE, int nN, int H, int D) {
    int e    = blockIdx.x * (blockDim.x >> 5) + (threadIdx.x >> 5);
    int lane = threadIdx.x & 31;
    if (e >= nE + nN) return;
    int s, d; edge_sd(ei, e, nE, s, d);
    float alpha[2];
    for (int h = 0; h < H; ++h) {
        float v = leaky(as_[(size_t)s * H + h] + ad_[(size_t)d * H + h]);
        float m = fdec(mx[(size_t)d * H + h]);
        alpha[h] = expf(v - m) / den[(size_t)d * H + h];
    }
    int HD = H * D;
    for (int j = lane; j < HD; j += 32)
        atomicAdd(&out[(size_t)d * HD + j], xl[(size_t)s * HD + j] * alpha[j / D]);
}

// ---------------------------------------------------------------------------
// buf = elu(buf + bias)  (in place)
// ---------------------------------------------------------------------------
__global__ void bias_elu_kernel(float* __restrict__ buf, const float* __restrict__ bias,
                                int n, int Cc) {
    int i = blockIdx.x * blockDim.x + threadIdx.x;
    if (i >= n * Cc) return;
    buf[i] = elu(buf[i] + bias[i % Cc]);
}

// ---------------------------------------------------------------------------
// global mean pool: accumulate + divide
// ---------------------------------------------------------------------------
__global__ void pool_accum_kernel(const float* __restrict__ h, const int* __restrict__ batch,
                                  float* __restrict__ sums, float* __restrict__ cnt,
                                  int nN, int Cc) {
    int node = blockIdx.x * (blockDim.x >> 5) + (threadIdx.x >> 5);
    int lane = threadIdx.x & 31;
    if (node >= nN) return;
    int b = batch[node];
    for (int j = lane; j < Cc; j += 32)
        atomicAdd(&sums[(size_t)b * Cc + j], h[(size_t)node * Cc + j]);
    if (lane == 0) atomicAdd(&cnt[b], 1.0f);
}

__global__ void pool_div_kernel(const float* __restrict__ sums, const float* __restrict__ cnt,
                                float* __restrict__ out, int G, int Cc) {
    int i = blockIdx.x * blockDim.x + threadIdx.x;
    if (i >= G * Cc) return;
    out[i] = sums[i] / fmaxf(cnt[i / Cc], 1.0f);
}

// ---------------------------------------------------------------------------
extern "C" void kernel_launch(void* const* d_in, const int* in_sizes, int n_in,
                              void* d_out, int out_size, void* d_ws, size_t ws_size,
                              hipStream_t stream) {
    const float* x     = (const float*)d_in[0];
    const int*   ei    = (const int*)  d_in[1];
    const int*   batch = (const int*)  d_in[2];
    const float* W1    = (const float*)d_in[3];
    const float* aS1   = (const float*)d_in[4];
    const float* aD1   = (const float*)d_in[5];
    const float* b1    = (const float*)d_in[6];
    const float* W2    = (const float*)d_in[7];
    const float* aS2   = (const float*)d_in[8];
    const float* aD2   = (const float*)d_in[9];
    const float* b2    = (const float*)d_in[10];
    float* out = (float*)d_out;

    const int nN = in_sizes[2];          // 50000 nodes
    const int nE = in_sizes[1] / 2;      // 1600000 edges
    const int ET = nE + nN;              // + self loops
    const int G  = out_size / OUT_CH;    // 64 graphs

    // ---- carve workspace (256B aligned) ----
    char* p = (char*)d_ws;
    auto carve = [&](size_t bytes) -> char* {
        char* q = p;
        p += (bytes + 255) & ~(size_t)255;
        return q;
    };
    _Float16* wt1  = (_Float16*)carve((size_t)C1 * IN_CH * 2);
    _Float16* wt2  = (_Float16*)carve((size_t)OUT_CH * C1 * 2);
    float*    xl1  = (float*)   carve((size_t)nN * C1 * 4);
    float*    as1  = (float*)   carve((size_t)nN * HEADS * 4);
    float*    ad1  = (float*)   carve((size_t)nN * HEADS * 4);
    unsigned* mx1  = (unsigned*)carve((size_t)nN * HEADS * 4);
    float*    den1 = (float*)   carve((size_t)nN * HEADS * 4);
    float*    agg1 = (float*)   carve((size_t)nN * C1 * 4);     // becomes h1 after elu
    float*    xl2  = (float*)   carve((size_t)nN * OUT_CH * 4);
    float*    as2  = (float*)   carve((size_t)nN * 4);
    float*    ad2  = (float*)   carve((size_t)nN * 4);
    unsigned* mx2  = (unsigned*)carve((size_t)nN * 4);
    float*    den2 = (float*)   carve((size_t)nN * 4);
    float*    agg2 = (float*)   carve((size_t)nN * OUT_CH * 4);
    float*    sums = (float*)   carve((size_t)G * OUT_CH * 4);
    float*    cnt  = (float*)   carve((size_t)G * 4);
    if ((size_t)(p - (char*)d_ws) > ws_size) return;  // insufficient scratch

    // ---- zero accumulators (ordered-uint max identity is 0u too) ----
    hipMemsetAsync(mx1,  0, (size_t)nN * HEADS * 4, stream);
    hipMemsetAsync(den1, 0, (size_t)nN * HEADS * 4, stream);
    hipMemsetAsync(agg1, 0, (size_t)nN * C1 * 4,    stream);
    hipMemsetAsync(mx2,  0, (size_t)nN * 4,         stream);
    hipMemsetAsync(den2, 0, (size_t)nN * 4,         stream);
    hipMemsetAsync(agg2, 0, (size_t)nN * OUT_CH * 4, stream);
    hipMemsetAsync(sums, 0, (size_t)G * OUT_CH * 4, stream);
    hipMemsetAsync(cnt,  0, (size_t)G * 4,          stream);

    // ---- convert / transpose weights to f16 ----
    conv_w_kernel<<<(IN_CH * OUT_CH + 255) / 256, 256, 0, stream>>>(W1, W2, wt1, wt2);

    const int gemmBlocks = (nN + 63) / 64;
    const int waveBlocksN  = (nN + 7) / 8;   // 8 waves/block, wave-per-node
    const int edgeThBlocks = (ET + 255) / 256;
    const int edgeWvBlocks = (ET + 7) / 8;   // wave-per-edge

    // ================= layer 1 =================
    gemm_wmma<IN_CH, C1><<<gemmBlocks, 128, 0, stream>>>(x, wt1, xl1, nN);
    attn_coef<<<waveBlocksN, 256, 0, stream>>>(xl1, aS1, aD1, as1, ad1, nN, HEADS, HID);
    edge_max_kernel<<<edgeThBlocks, 256, 0, stream>>>(ei, as1, ad1, mx1, nE, nN, HEADS);
    edge_den_kernel<<<edgeThBlocks, 256, 0, stream>>>(ei, as1, ad1, mx1, den1, nE, nN, HEADS);
    edge_scatter_kernel<<<edgeWvBlocks, 256, 0, stream>>>(ei, as1, ad1, mx1, den1,
                                                          xl1, agg1, nE, nN, HEADS, HID);
    bias_elu_kernel<<<((size_t)nN * C1 + 255) / 256, 256, 0, stream>>>(agg1, b1, nN, C1);

    // ================= layer 2 =================
    gemm_wmma<C1, OUT_CH><<<gemmBlocks, 128, 0, stream>>>(agg1, wt2, xl2, nN);
    attn_coef<<<waveBlocksN, 256, 0, stream>>>(xl2, aS2, aD2, as2, ad2, nN, 1, OUT_CH);
    edge_max_kernel<<<edgeThBlocks, 256, 0, stream>>>(ei, as2, ad2, mx2, nE, nN, 1);
    edge_den_kernel<<<edgeThBlocks, 256, 0, stream>>>(ei, as2, ad2, mx2, den2, nE, nN, 1);
    edge_scatter_kernel<<<edgeWvBlocks, 256, 0, stream>>>(ei, as2, ad2, mx2, den2,
                                                          xl2, agg2, nE, nN, 1, OUT_CH);
    bias_elu_kernel<<<((size_t)nN * OUT_CH + 255) / 256, 256, 0, stream>>>(agg2, b2, nN, OUT_CH);

    // ================= pooling =================
    pool_accum_kernel<<<waveBlocksN, 256, 0, stream>>>(agg2, batch, sums, cnt, nN, OUT_CH);
    pool_div_kernel<<<(G * OUT_CH + 255) / 256, 256, 0, stream>>>(sums, cnt, out, G, OUT_CH);
}
